// PointnetSAModuleFSMSGWithImage_49469433315934
// MI455X (gfx1250) — matrix operand order repaired and checked by reading, not compiled, not run on hardware
//
#include <hip/hip_runtime.h>
#include <hip/hip_bf16.h>

// ---------------------------------------------------------------------------
// PointNet++ SA-MSG module for MI455X (gfx1250, wave32, WMMA).
// All 1x1-conv layers run on v_wmma_f32_16x16x32_f16 (fp16 in, fp32 acc).
// Grouped activations live entirely in LDS; pooled features ping-pong through
// f16 workspace buffers between GEMM stages.
// ---------------------------------------------------------------------------

#define BATCH   4
#define NPTS    8192
#define NFEAT   64
#define NPOINT  2048

typedef __attribute__((ext_vector_type(16))) _Float16 v16h;
typedef __attribute__((ext_vector_type(8)))  _Float16 v8h;
typedef __attribute__((ext_vector_type(8)))  float    v8f;
typedef __attribute__((ext_vector_type(4)))  float    v4f;

// ---- workspace layout (bytes) ----
constexpr size_t OFF_DISTS  = 0;                          // B*NPTS f32      = 131072
constexpr size_t OFF_SIDX   = OFF_DISTS  + 131072;        // B*NPOINT i32    = 32768
constexpr size_t OFF_IDX0   = OFF_SIDX   + 32768;         // B*NPOINT*16 i32 = 524288
constexpr size_t OFF_VALID0 = OFF_IDX0   + 524288;        // B*NPOINT i32    = 32768
constexpr size_t OFF_IDX1   = OFF_VALID0 + 32768;         // B*NPOINT*32 i32 = 1048576
constexpr size_t OFF_VALID1 = OFF_IDX1   + 1048576;       // B*NPOINT i32    = 32768
constexpr size_t OFF_POOLED = OFF_VALID1 + 32768;         // B*NPOINT*256 f16= 4194304
constexpr size_t OFF_AGGOUT = OFF_POOLED + 4194304;       // B*NPOINT*256 f16= 4194304
constexpr size_t OFF_H      = OFF_AGGOUT + 4194304;       // B*NPOINT*128 f16= 2097152
constexpr size_t OFF_WPACK  = OFF_H      + 2097152;       // 141312 halves   = 282624
// total ~12.6 MB of d_ws

// packed weight offsets (in halves), rows = cout, row stride = cin padded to 32
constexpr size_t WOFF0 = 0;        // scale0 L1: 64 x 96  (cin 67)
constexpr size_t WOFF1 = 6144;     // scale0 L2: 64 x 64
constexpr size_t WOFF2 = 10240;    // scale0 L3: 128 x 64
constexpr size_t WOFF3 = 18432;    // scale1 L1: 64 x 96  (cin 67)
constexpr size_t WOFF4 = 24576;    // scale1 L2: 96 x 64
constexpr size_t WOFF5 = 30720;    // scale1 L3: 128 x 96
constexpr size_t WOFF6 = 43008;    // agg    : 256 x 256
constexpr size_t WOFF7 = 108544;   // conf   : 128 x 256

// ---------------------------------------------------------------------------
// Weight prep: fp32 -> f16, pad cin to multiple of 32 with zeros.
// ---------------------------------------------------------------------------
struct LayerDesc { const float* W; int cout; int cin; int cin_pad; unsigned dst_off; };
struct PrepArgs  { LayerDesc l[8]; };

__global__ void prep_weights_kernel(PrepArgs a, _Float16* __restrict__ wbase) {
  LayerDesc d = a.l[blockIdx.x];
  const int total = d.cout * d.cin_pad;
  for (int i = threadIdx.x; i < total; i += blockDim.x) {
    const int r = i / d.cin_pad, c = i % d.cin_pad;
    wbase[d.dst_off + i] = (c < d.cin) ? (_Float16)d.W[r * d.cin + c] : (_Float16)0.0f;
  }
}

// ---------------------------------------------------------------------------
// Farthest point sampling: one block per batch, serial over NPOINT steps,
// 1024-thread LDS argmax per step. Also emits new_xyz (output 0).
// ---------------------------------------------------------------------------
__global__ __launch_bounds__(1024) void fps_kernel(
    const float* __restrict__ xyz, float* __restrict__ dists,
    int* __restrict__ sidx, float* __restrict__ newxyz) {
  __shared__ float sval[1024];
  __shared__ int   sarg[1024];
  __shared__ int   far_s;
  const int b = blockIdx.x, tid = threadIdx.x;
  const float* xb = xyz + (size_t)b * NPTS * 3;
  float* db = dists + (size_t)b * NPTS;
  for (int n = tid; n < NPTS; n += 1024) db[n] = 1e10f;
  if (tid == 0) far_s = 0;
  __syncthreads();
  for (int i = 0; i < NPOINT; i++) {
    const int far = far_s;
    const float fx = xb[far*3+0], fy = xb[far*3+1], fz = xb[far*3+2];
    if (tid == 0) {
      sidx[(size_t)b*NPOINT + i] = far;
      float* nz = newxyz + ((size_t)b*NPOINT + i)*3;
      nz[0] = fx; nz[1] = fy; nz[2] = fz;
    }
    float best = -1.0f; int bi = 0;
    for (int n = tid; n < NPTS; n += 1024) {
      const float dx = xb[n*3+0]-fx, dy = xb[n*3+1]-fy, dz = xb[n*3+2]-fz;
      float d = dx*dx + dy*dy + dz*dz;
      d = fminf(db[n], d);
      db[n] = d;
      if (d > best) { best = d; bi = n; }
    }
    sval[tid] = best; sarg[tid] = bi;
    __syncthreads();
    for (int off = 512; off > 0; off >>= 1) {
      if (tid < off) {
        const float ov = sval[tid+off]; const int oa = sarg[tid+off];
        if (ov > sval[tid] || (ov == sval[tid] && oa < sarg[tid])) { sval[tid] = ov; sarg[tid] = oa; }
      }
      __syncthreads();
    }
    if (tid == 0) far_s = sarg[0];
    __syncthreads();
  }
}

// ---------------------------------------------------------------------------
// Ball query: one thread per (b,s). First-nsample indices (index order ==
// sorted order), pad with first hit / 0; valid = any hit (all the reference
// uses of cnt reduce to cnt>0).
// ---------------------------------------------------------------------------
__global__ void ball_query_kernel(
    const float* __restrict__ xyz, const float* __restrict__ newxyz,
    int* __restrict__ idxb, int* __restrict__ validb, float r2, int K) {
  const int t = blockIdx.x * blockDim.x + threadIdx.x;
  if (t >= BATCH * NPOINT) return;
  const int b = t / NPOINT;
  const float cx = newxyz[t*3+0], cy = newxyz[t*3+1], cz = newxyz[t*3+2];
  const float* xb = xyz + (size_t)b * NPTS * 3;
  int* ob = idxb + (size_t)t * K;
  int cnt = 0;
  for (int n = 0; n < NPTS && cnt < K; n++) {
    const float dx = xb[n*3+0]-cx, dy = xb[n*3+1]-cy, dz = xb[n*3+2]-cz;
    if (dx*dx + dy*dy + dz*dz < r2) ob[cnt++] = n;
  }
  validb[t] = (cnt > 0) ? 1 : 0;
  const int fill = (cnt > 0) ? ob[0] : 0;
  for (int j = cnt; j < K; j++) ob[j] = fill;
}

// ---------------------------------------------------------------------------
// WMMA layer: D(16x16 f32) = A(16x32 f16 weights) x B(32x16 f16 activations),
// accumulated over KB k-blocks, epilogue relu(g*acc+b) stored as f16.
// Lane packing: A/B use contiguous k (lane half h holds k 16h..16h+15), C/D per CDNA5 ISA 7.12.2.
// ---------------------------------------------------------------------------
template<int KB>
__device__ __forceinline__ void wmma_layer(
    const _Float16* __restrict__ Xin, int in_stride,
    _Float16* __restrict__ Yout, int out_stride,
    const _Float16* __restrict__ Wp,
    const float* __restrict__ gg, const float* __restrict__ bb,
    int cout, int n_mtiles) {
  const int wave = threadIdx.x >> 5;
  const int lane = threadIdx.x & 31;
  const int lhi = lane >> 4;          // half-wave select
  const int llo = lane & 15;
  const int nct = cout >> 4;
  const int total = nct * n_mtiles;   // always a multiple of 8 here
  for (int item = wave; item < total; item += 8) {
    const int ct = item % nct;
    const int mt = item / nct;
    v8f acc = {};
#pragma unroll
    for (int kb = 0; kb < KB; kb++) {
      const int k0 = kb * 32;
      // A: row = cout-in-tile = llo; lane<16 holds K k0..k0+15,
      //    lane>=16 holds K k0+16..31
      const _Float16* ap = Wp + (size_t)(ct*16 + llo) * (KB*32) + k0 + lhi*16;
      const v8h alo = *(const v8h*)ap;
      const v8h ahi = *(const v8h*)(ap + 8);
      v16h a;
#pragma unroll
      for (int i = 0; i < 8; i++) { a[i] = alo[i]; a[8+i] = ahi[i]; }
      // B: col = m-in-tile = llo; lane<16 holds K k0..k0+15, lane>=16 K k0+16..31
      const v16h bf = *(const v16h*)(Xin + (size_t)(mt*16 + llo) * in_stride + k0 + lhi*16);
      acc = __builtin_amdgcn_wmma_f32_16x16x32_f16(false, a, false, bf,
                                                   (short)0, acc, false, false);
    }
    // D: VGPR v, lane<16 -> (cout ct*16+v, m llo); lane>=16 -> (ct*16+8+v, llo)
    const int m  = mt*16 + llo;
    const int cb = ct*16 + lhi*8;
    const v4f g0 = *(const v4f*)(gg + cb);
    const v4f g1 = *(const v4f*)(gg + cb + 4);
    const v4f b0 = *(const v4f*)(bb + cb);
    const v4f b1 = *(const v4f*)(bb + cb + 4);
    v8h outv;
#pragma unroll
    for (int v = 0; v < 4; v++) {
      outv[v]   = (_Float16)fmaxf(acc[v]   * g0[v] + b0[v], 0.0f);
      outv[v+4] = (_Float16)fmaxf(acc[v+4] * g1[v] + b1[v], 0.0f);
    }
    *(v8h*)(Yout + (size_t)m * out_stride + cb) = outv;
  }
}

// ---------------------------------------------------------------------------
// Fused grouping + 3-layer MLP + max-pool per scale. Block = 256 threads
// (8 wave32), M = 128 grouped columns resident in LDS.
//   scale0: K=16, layers 67->64->64->128   (COUT2=64,  KB3=2)
//   scale1: K=32, layers 67->64->96->128   (COUT2=96,  KB3=3)
// ---------------------------------------------------------------------------
template<int K, int COUT2, int KB3>
__global__ __launch_bounds__(256) void group_mlp_kernel(
    const float* __restrict__ xyz, const float* __restrict__ feat,
    const float* __restrict__ newxyz,
    const int* __restrict__ idxb, const int* __restrict__ validb,
    const _Float16* __restrict__ W1, const float* __restrict__ g1, const float* __restrict__ b1,
    const _Float16* __restrict__ W2, const float* __restrict__ g2, const float* __restrict__ b2,
    const _Float16* __restrict__ W3, const float* __restrict__ g3, const float* __restrict__ b3,
    _Float16* __restrict__ pooled, int coffset) {
  constexpr int M  = 128;
  constexpr int SB = M / K;           // sample points per block
  __shared__ __align__(32) _Float16 Xs[M * 96];   // 24 KB
  __shared__ __align__(32) _Float16 Ys[M * 128];  // 32 KB
  const int b = blockIdx.y, sblk = blockIdx.x, tid = threadIdx.x;

  // ---- grouping: X[mi][c] = concat(xyz[idx]-new_xyz, features[:,idx], 0pad)
  {
    const int mi = tid >> 1, half = tid & 1;      // two threads per column
    const int sl = mi / K, kk = mi % K;
    const int s  = sblk * SB + sl;
    const int gi = idxb[((size_t)b*NPOINT + s)*K + kk];
    const float* pp = xyz    + ((size_t)b*NPTS   + gi)*3;
    const float* nz = newxyz + ((size_t)b*NPOINT + s )*3;
    _Float16* xr = &Xs[mi*96 + half*48];
    for (int c0 = 0; c0 < 48; c0++) {
      const int c = half*48 + c0;
      float v;
      if (c < 3)       v = pp[c] - nz[c];
      else if (c < 67) v = feat[((size_t)b*NFEAT + (c-3))*NPTS + gi];
      else             v = 0.0f;
      xr[c0] = (_Float16)v;
    }
  }
  __syncthreads();
  wmma_layer<3>  (Xs, 96,  Ys, 128, W1, g1, b1, 64,    M/16);  // 67(pad96)->64
  __syncthreads();
  wmma_layer<2>  (Ys, 128, Xs, 96,  W2, g2, b2, COUT2, M/16);  // 64->COUT2
  __syncthreads();
  wmma_layer<KB3>(Xs, 96,  Ys, 128, W3, g3, b3, 128,   M/16);  // COUT2->128
  __syncthreads();
  // ---- max-pool over K (relu output >= 0 so 0-init is exact) + validity
  for (int o = tid; o < SB * 128; o += 256) {
    const int sl = o >> 7, c = o & 127;
    const int s  = sblk * SB + sl;
    float mx = 0.0f;
    for (int kk = 0; kk < K; kk++)
      mx = fmaxf(mx, (float)Ys[(sl*K + kk)*128 + c]);
    if (!validb[(size_t)b*NPOINT + s]) mx = 0.0f;
    pooled[((size_t)b*NPOINT + s)*256 + coffset + c] = (_Float16)mx;
  }
}

// ---------------------------------------------------------------------------
// 1D MLP layer (cin = 256): block handles (batch, 16 sample points), waves own
// cout tiles. Optionally also emits fp32 output in [b][cout][NPOINT] order
// (this is the module's new_features output for the agg layer).
// ---------------------------------------------------------------------------
__global__ __launch_bounds__(256) void mlp1d_kernel(
    const _Float16* __restrict__ Xg,   // [B][NPOINT][256] f16
    const _Float16* __restrict__ Wp,   // [cout][256] f16
    const float* __restrict__ gg, const float* __restrict__ bb,
    _Float16* __restrict__ Yg, int cout,   // [B][NPOINT][cout] f16
    float* __restrict__ outf) {            // optional [B][cout][NPOINT] f32
  constexpr int KB = 8, CIN = 256;
  __shared__ __align__(32) _Float16 Xt[16 * CIN];   // 8 KB
  const int b = blockIdx.y, mblk = blockIdx.x, tid = threadIdx.x;
  {
    const uint4* s4 = (const uint4*)(Xg + ((size_t)b*NPOINT + mblk*16)*CIN);
    uint4* d4 = (uint4*)Xt;
    for (int i = tid; i < 16*CIN/8; i += 256) d4[i] = s4[i];
  }
  __syncthreads();
  const int wave = tid >> 5, lane = tid & 31;
  const int lhi = lane >> 4, llo = lane & 15;
  const int nct = cout >> 4;
  for (int ct = wave; ct < nct; ct += 8) {
    v8f acc = {};
#pragma unroll
    for (int kb = 0; kb < KB; kb++) {
      const int k0 = kb * 32;
      const _Float16* ap = Wp + (size_t)(ct*16 + llo)*CIN + k0 + lhi*16;
      const v8h alo = *(const v8h*)ap;
      const v8h ahi = *(const v8h*)(ap + 8);
      v16h a;
#pragma unroll
      for (int i = 0; i < 8; i++) { a[i] = alo[i]; a[8+i] = ahi[i]; }
      const v16h bf = *(const v16h*)(Xt + (size_t)llo*CIN + k0 + lhi*16);
      acc = __builtin_amdgcn_wmma_f32_16x16x32_f16(false, a, false, bf,
                                                   (short)0, acc, false, false);
    }
    const int s  = mblk*16 + llo;
    const int cb = ct*16 + lhi*8;
    const v4f g0 = *(const v4f*)(gg + cb);
    const v4f g1 = *(const v4f*)(gg + cb + 4);
    const v4f b0 = *(const v4f*)(bb + cb);
    const v4f b1 = *(const v4f*)(bb + cb + 4);
    v8f vals;
#pragma unroll
    for (int v = 0; v < 4; v++) {
      vals[v]   = fmaxf(acc[v]   * g0[v] + b0[v], 0.0f);
      vals[v+4] = fmaxf(acc[v+4] * g1[v] + b1[v], 0.0f);
    }
    v8h outv;
#pragma unroll
    for (int v = 0; v < 8; v++) outv[v] = (_Float16)vals[v];
    *(v8h*)(Yg + ((size_t)b*NPOINT + s)*cout + cb) = outv;
    if (outf) {                                  // uniform branch, one clause
      float* op = outf + ((size_t)b*cout + cb)*NPOINT + s;
#pragma unroll
      for (int v = 0; v < 8; v++) op[(size_t)v*NPOINT] = vals[v];
    }
  }
}

// ---------------------------------------------------------------------------
// Final confidence head: scores[b,s] = final_w . h[b,s,:] + final_b
// ---------------------------------------------------------------------------
__global__ void scores_kernel(const _Float16* __restrict__ h,
                              const float* __restrict__ fw, const float* __restrict__ fb,
                              float* __restrict__ out) {
  const int t = blockIdx.x * blockDim.x + threadIdx.x;
  if (t >= BATCH * NPOINT) return;
  float acc = fb[0];
  const _Float16* hp = h + (size_t)t * 128;
  for (int c = 0; c < 128; c++) acc += fw[c] * (float)hp[c];
  out[t] = acc;
}

// ---------------------------------------------------------------------------
// Launch. d_in flattened setup_inputs() order:
//  0 xyz, 1 features,
//  2..10  mlp_params[0]: (W,g,b) x 3        -- 67->64, 64->64, 64->128
//  11..19 mlp_params[1]: (W,g,b) x 3        -- 67->64, 64->96, 96->128
//  20..22 agg_params: W(256x256), g, b
//  23..25 conf layers: W(128x256), g, b ; 26 final_w(1x128) ; 27 final_b(1)
// d_out = [new_xyz B*2048*3 | new_features B*256*2048 | scores B*2048] f32.
// ---------------------------------------------------------------------------
extern "C" void kernel_launch(void* const* d_in, const int* in_sizes, int n_in,
                              void* d_out, int out_size, void* d_ws, size_t ws_size,
                              hipStream_t stream) {
  (void)in_sizes; (void)n_in; (void)out_size; (void)ws_size;
  const float* xyz  = (const float*)d_in[0];
  const float* feat = (const float*)d_in[1];

  char* ws = (char*)d_ws;
  float*     dists  = (float*)(ws + OFF_DISTS);
  int*       sidx   = (int*)  (ws + OFF_SIDX);
  int*       idx0   = (int*)  (ws + OFF_IDX0);
  int*       valid0 = (int*)  (ws + OFF_VALID0);
  int*       idx1   = (int*)  (ws + OFF_IDX1);
  int*       valid1 = (int*)  (ws + OFF_VALID1);
  _Float16*  pooled = (_Float16*)(ws + OFF_POOLED);
  _Float16*  aggout = (_Float16*)(ws + OFF_AGGOUT);
  _Float16*  hbuf   = (_Float16*)(ws + OFF_H);
  _Float16*  wpack  = (_Float16*)(ws + OFF_WPACK);

  float* out     = (float*)d_out;
  float* newxyz  = out;                                   // [B][NPOINT][3]
  float* newfeat = out + (size_t)BATCH*NPOINT*3;          // [B][256][NPOINT]
  float* scores  = newfeat + (size_t)BATCH*256*NPOINT;    // [B][NPOINT]

  // --- pack weights to f16 (padded) ---
  PrepArgs pa;
  pa.l[0] = { (const float*)d_in[2],  64, 67,  96, (unsigned)WOFF0 };
  pa.l[1] = { (const float*)d_in[5],  64, 64,  64, (unsigned)WOFF1 };
  pa.l[2] = { (const float*)d_in[8], 128, 64,  64, (unsigned)WOFF2 };
  pa.l[3] = { (const float*)d_in[11], 64, 67,  96, (unsigned)WOFF3 };
  pa.l[4] = { (const float*)d_in[14], 96, 64,  64, (unsigned)WOFF4 };
  pa.l[5] = { (const float*)d_in[17],128, 96,  96, (unsigned)WOFF5 };
  pa.l[6] = { (const float*)d_in[20],256, 256, 256,(unsigned)WOFF6 };
  pa.l[7] = { (const float*)d_in[23],128, 256, 256,(unsigned)WOFF7 };
  prep_weights_kernel<<<8, 256, 0, stream>>>(pa, wpack);

  // --- FPS (also writes new_xyz output) ---
  fps_kernel<<<BATCH, 1024, 0, stream>>>(xyz, dists, sidx, newxyz);

  // --- ball queries ---
  const int nq = BATCH * NPOINT;
  ball_query_kernel<<<(nq+255)/256, 256, 0, stream>>>(xyz, newxyz, idx0, valid0, 0.25f, 16);
  ball_query_kernel<<<(nq+255)/256, 256, 0, stream>>>(xyz, newxyz, idx1, valid1, 1.00f, 32);

  // --- fused group + MLP + maxpool per scale ---
  group_mlp_kernel<16, 64, 2><<<dim3(NPOINT/8, BATCH), 256, 0, stream>>>(
      xyz, feat, newxyz, idx0, valid0,
      wpack + WOFF0, (const float*)d_in[3],  (const float*)d_in[4],
      wpack + WOFF1, (const float*)d_in[6],  (const float*)d_in[7],
      wpack + WOFF2, (const float*)d_in[9],  (const float*)d_in[10],
      pooled, /*coffset=*/0);
  group_mlp_kernel<32, 96, 3><<<dim3(NPOINT/4, BATCH), 256, 0, stream>>>(
      xyz, feat, newxyz, idx1, valid1,
      wpack + WOFF3, (const float*)d_in[12], (const float*)d_in[13],
      wpack + WOFF4, (const float*)d_in[15], (const float*)d_in[16],
      wpack + WOFF5, (const float*)d_in[18], (const float*)d_in[19],
      pooled, /*coffset=*/128);

  // --- agg MLP (emits new_features fp32) and conf MLP ---
  mlp1d_kernel<<<dim3(NPOINT/16, BATCH), 256, 0, stream>>>(
      pooled, wpack + WOFF6, (const float*)d_in[21], (const float*)d_in[22],
      aggout, 256, newfeat);
  mlp1d_kernel<<<dim3(NPOINT/16, BATCH), 256, 0, stream>>>(
      aggout, wpack + WOFF7, (const float*)d_in[24], (const float*)d_in[25],
      hbuf, 128, nullptr);

  // --- final scores ---
  scores_kernel<<<(nq+255)/256, 256, 0, stream>>>(
      hbuf, (const float*)d_in[26], (const float*)d_in[27], scores);
}
